// MambaBlock_31129922961502
// MI455X (gfx1250) — matrix operand order repaired and checked by reading, not compiled, not run on hardware
//
#include <hip/hip_runtime.h>
#include <hip/hip_bf16.h>
#include <stdint.h>

// ---------------------------------------------------------------------------
// Mamba block forward for MI455X (gfx1250, wave32, WMMA + TDM).
// Big GEMMs: 128x128 block tiles, bf16 WMMA, double-buffered LDS staged by the
// Tensor Data Mover (tensor_load_to_lds + s_wait_tensorcnt), TDM pads LDS rows
// to 80B so 16-lane fragment reads are bank-conflict-free.
// Selective scan: 16 lanes per (batch, d_inner) channel, one state per lane,
// cross-lane reduction via __shfl_xor, fused silu(z) gating.
// ---------------------------------------------------------------------------

#define D_MODEL 1024
#define D_STATE 16
#define D_CONV  4
#define D_INNER 2048
#define DT_RANK 64
#define BATCH   2
#define SEQ     1024
#define BT      (BATCH * SEQ)          // 2048 rows

typedef __attribute__((ext_vector_type(16))) __bf16          v16bf;
typedef __attribute__((ext_vector_type(8)))  float           v8f;
typedef __attribute__((ext_vector_type(8)))  unsigned short  ush8;
typedef __attribute__((ext_vector_type(4)))  unsigned int    u32x4;
typedef __attribute__((ext_vector_type(8)))  int             i32x8;
typedef __attribute__((ext_vector_type(4)))  int             i32x4;

#ifndef __has_builtin
#define __has_builtin(x) 0
#endif
#if __has_builtin(__builtin_amdgcn_tensor_load_to_lds) && \
    __has_builtin(__builtin_amdgcn_s_wait_tensorcnt)
#define MAMBA_USE_TDM 1
#else
#define MAMBA_USE_TDM 0
#endif

// LDS tile geometry: 128 rows x 32 bf16, row stride padded to 40 halves (80B).
#define TILE_ROWS   128
#define TILE_K      32
#define LDS_STRIDE  40
#define TILE_SHORTS (TILE_ROWS * LDS_STRIDE)     // 5120 halves = 10240 B

// ------------------------------ helpers ------------------------------------

__device__ __forceinline__ unsigned short f32_to_bf16_bits(float f) {
  unsigned int u = __float_as_uint(f);
  unsigned int rnd = 0x7FFFu + ((u >> 16) & 1u);   // round-to-nearest-even
  u += rnd;
  return (unsigned short)(u >> 16);
}

__device__ __forceinline__ float bf16_bits_to_f32(unsigned short h) {
  return __uint_as_float(((unsigned int)h) << 16);
}

__device__ __forceinline__ float silu(float x) {
  return x / (1.0f + __expf(-x));
}

__device__ __forceinline__ float softplus_stable(float x) {
  return fmaxf(x, 0.0f) + log1pf(__expf(-fabsf(x)));
}

// CDNA5 16-bit A-matrix 16x32 layout (ISA 7.12.2):
//   lanes 0-15 : row M = lane,     VGPR0-3: K=0..7,  VGPR4-7: K=16..23
//   lanes 16-31: row M = lane-16,  VGPR0-3: K=8..15, VGPR4-7: K=24..31
__device__ __forceinline__ v16bf frag_a(const unsigned short* base, int stride,
                                        int row0, int k0, int lane) {
  int r  = row0 + (lane & 15);
  int kh = ((lane >> 4) & 1) << 3;                 // 0 or 8
  const unsigned short* p = base + (size_t)r * stride + k0 + kh;
  union { ush8 u[2]; v16bf v; } x;
  x.u[0] = *(const ush8*)(p);                      // K = kh .. kh+7
  x.u[1] = *(const ush8*)(p + 16);                 // K = 16+kh .. 16+kh+7
  return x.v;
}

// B-matrix 32x16 (K-major per column): lane n holds column N,
// lanes 0-15 -> K=0..7,16..23, lanes 16-31 -> K=8..15,24..31. Source row-major W[N][K].
__device__ __forceinline__ v16bf frag_b(const unsigned short* base, int stride,
                                        int row0, int k0, int lane) {
  int r  = row0 + (lane & 15);
  int kh = ((lane >> 4) & 1) << 3;                 // 0 or 8
  const unsigned short* p = base + (size_t)r * stride + k0 + kh;
  union { ush8 u[2]; v16bf v; } x;
  x.u[0] = *(const ush8*)(p);
  x.u[1] = *(const ush8*)(p + 16);
  return x.v;
}

__device__ __forceinline__ v8f wmma_bf16(v16bf a, v16bf b, v8f c) {
  return __builtin_amdgcn_wmma_f32_16x16x32_bf16(
      false, a, false, b, (short)0, c, false, false);
}

// TDM: move one 128x32 bf16 tile (row-major, leading dim `ld` halves) from
// global to LDS at byte offset lds_off, padding 16B after every 64B so the
// LDS row stride is 80B. One instruction, tracked by TENSORcnt.
__device__ __forceinline__ void tdm_load_tile(const unsigned short* g, int ld,
                                              int row0, int k0,
                                              unsigned lds_off) {
#if MAMBA_USE_TDM
  unsigned long long ga = (unsigned long long)(size_t)g +
      ((unsigned long long)(unsigned)row0 * (unsigned)ld + (unsigned)k0) * 2ull;
  u32x4 g0;
  g0[0] = 1u;                                   // count=1, user mode
  g0[1] = lds_off;                              // lds_addr (bytes)
  g0[2] = (unsigned)ga;                         // global_addr[31:0]
  g0[3] = (unsigned)((ga >> 32) & 0x01FFFFFFu)  // global_addr[56:32]
        | (2u << 30);                           // type = 2 (image)
  i32x8 g1;
  g1[0] = (int)((1u << 16)      // data_size = 2 bytes
              | (1u << 20)      // pad_enable
              | (3u << 22)      // pad_interval: 16 DWORDs (64B)
              | (3u << 25));    // pad_amount: 4 DWORDs (16B)
  unsigned td0 = 1u << 30, td1 = 1u << 30;      // huge tensor dims (no OOB)
  g1[1] = (int)((td0 & 0xFFFFu) << 16);
  g1[2] = (int)(((td0 >> 16) & 0xFFFFu) | ((td1 & 0xFFFFu) << 16));
  g1[3] = (int)(((td1 >> 16) & 0xFFFFu) | ((unsigned)TILE_K << 16)); // tile_dim0=32
  g1[4] = (int)TILE_ROWS;                       // tile_dim1=128, tile_dim2=0
  unsigned long long s0 = (unsigned)ld;         // tensor_dim0_stride = ld
  g1[5] = (int)(s0 & 0xFFFFFFFFu);
  g1[6] = (int)((s0 >> 32) & 0xFFFFu);          // + tensor_dim1_stride lo = 0
  g1[7] = 0;
  i32x4 gz = {0, 0, 0, 0};
#if __clang_major__ >= 23
  i32x8 gz8 = {0, 0, 0, 0, 0, 0, 0, 0};
  __builtin_amdgcn_tensor_load_to_lds(g0, g1, gz, gz, gz8, 0);
#else
  __builtin_amdgcn_tensor_load_to_lds(g0, g1, gz, gz, 0);
#endif
#else
  (void)g; (void)ld; (void)row0; (void)k0; (void)lds_off;
#endif
}

// Fallback cooperative copy (256 threads): 128x32 halves -> LDS stride 40.
__device__ __forceinline__ void load_tile_direct(const unsigned short* g, int ld,
                                                 int row0, int k0,
                                                 unsigned short* lds) {
  #pragma unroll
  for (int j = 0; j < 2; ++j) {
    int c   = threadIdx.x * 2 + j;               // 0..511 chunks of 16B
    int row = c >> 2;
    int cc  = c & 3;
    const ush8* src = (const ush8*)(g + (size_t)(row0 + row) * ld + k0 + cc * 8);
    *(ush8*)(lds + row * LDS_STRIDE + cc * 8) = *src;
  }
}

// ------------------------------ kernels ------------------------------------

__global__ void cvt_f32_bf16_kernel(const float* __restrict__ src,
                                    unsigned short* __restrict__ dst, int n) {
  int i = blockIdx.x * blockDim.x + threadIdx.x;
  if (i < n) dst[i] = f32_to_bf16_bits(src[i]);
}

__device__ float block_reduce_sum_256(float v) {
  __shared__ float buf[8];
  __shared__ float result;
  #pragma unroll
  for (int off = 16; off > 0; off >>= 1) v += __shfl_xor(v, off, 32);
  int wid = threadIdx.x >> 5, lane = threadIdx.x & 31;
  __syncthreads();
  if (lane == 0) buf[wid] = v;
  __syncthreads();
  if (threadIdx.x == 0) {
    float r = 0.0f;
    #pragma unroll
    for (int i = 0; i < 8; ++i) r += buf[i];
    result = r;
  }
  __syncthreads();
  return result;
}

// LayerNorm over D_MODEL=1024, one row per 256-thread block; output bf16.
__global__ void ln_kernel(const float* __restrict__ x,
                          const float* __restrict__ w,
                          const float* __restrict__ b,
                          unsigned short* __restrict__ xn) {
  int row = blockIdx.x;
  const float* xr = x + (size_t)row * D_MODEL;
  float vals[4];
  float s = 0.0f;
  #pragma unroll
  for (int i = 0; i < 4; ++i) {
    vals[i] = xr[threadIdx.x + 256 * i];
    s += vals[i];
  }
  float mu = block_reduce_sum_256(s) * (1.0f / D_MODEL);
  float vs = 0.0f;
  #pragma unroll
  for (int i = 0; i < 4; ++i) {
    float d = vals[i] - mu;
    vs += d * d;
  }
  float var = block_reduce_sum_256(vs) * (1.0f / D_MODEL);
  float rstd = rsqrtf(var + 1e-5f);
  #pragma unroll
  for (int i = 0; i < 4; ++i) {
    int c = threadIdx.x + 256 * i;
    float o = (vals[i] - mu) * rstd * w[c] + b[c];
    xn[(size_t)row * D_MODEL + c] = f32_to_bf16_bits(o);
  }
}

// ---- Big GEMM: C[M,N] = A[M,K]*B[N,K]^T, 128x128 block tile, 8 waves. ----
// Wave w: 32-row strip (w>>1), 64-col strip (w&1) -> 2x4 WMMA tiles.
// mode 0: plain   mode 1: softplus(acc+bias[n])   mode 2: acc+resid[m,n]
__global__ void __launch_bounds__(256)
gemm_bf16_lds_kernel(const unsigned short* __restrict__ A, int lda,
                     const unsigned short* __restrict__ B, int ldb,
                     float* __restrict__ C, int ldc,
                     int K, int mode,
                     const float* __restrict__ bias,
                     const float* __restrict__ resid) {
  __shared__ unsigned short smem[2][2][TILE_SHORTS];   // [buf][A/B][tile]
  int lane = threadIdx.x & 31;
  int wid  = threadIdx.x >> 5;
  int wm   = wid >> 1;                 // 0..3
  int wn   = wid & 1;                  // 0..1
  int n0 = blockIdx.x * 128;
  int m0 = blockIdx.y * 128;
  v8f acc[2][4] = {};

#if MAMBA_USE_TDM
  unsigned ldsBase = (unsigned)(size_t)&smem[0][0][0];
  const unsigned TILE_BYTES = TILE_SHORTS * 2;       // 10240
  if (wid == 0) {
    tdm_load_tile(A, lda, m0, 0, ldsBase);
    tdm_load_tile(B, ldb, n0, 0, ldsBase + TILE_BYTES);
  }
  int buf = 0;
  for (int k0 = 0; k0 < K; k0 += TILE_K) {
    if (wid == 0) __builtin_amdgcn_s_wait_tensorcnt(0);
    __syncthreads();                                  // tile k0 visible to all
    if (k0 + TILE_K < K && wid == 0) {                // prefetch next tile
      unsigned nb = ldsBase + (unsigned)(buf ^ 1) * 2u * TILE_BYTES;
      tdm_load_tile(A, lda, m0, k0 + TILE_K, nb);
      tdm_load_tile(B, ldb, n0, k0 + TILE_K, nb + TILE_BYTES);
    }
    const unsigned short* As = &smem[buf][0][0];
    const unsigned short* Bs = &smem[buf][1][0];
    v16bf a0 = frag_a(As, LDS_STRIDE, wm * 32,      0, lane);
    v16bf a1 = frag_a(As, LDS_STRIDE, wm * 32 + 16, 0, lane);
    #pragma unroll
    for (int ni = 0; ni < 4; ++ni) {
      v16bf bfv = frag_b(Bs, LDS_STRIDE, wn * 64 + ni * 16, 0, lane);
      acc[0][ni] = wmma_bf16(a0, bfv, acc[0][ni]);
      acc[1][ni] = wmma_bf16(a1, bfv, acc[1][ni]);
    }
    buf ^= 1;
  }
#else
  for (int k0 = 0; k0 < K; k0 += TILE_K) {
    __syncthreads();
    load_tile_direct(A, lda, m0, k0, &smem[0][0][0]);
    load_tile_direct(B, ldb, n0, k0, &smem[0][1][0]);
    __syncthreads();
    const unsigned short* As = &smem[0][0][0];
    const unsigned short* Bs = &smem[0][1][0];
    v16bf a0 = frag_a(As, LDS_STRIDE, wm * 32,      0, lane);
    v16bf a1 = frag_a(As, LDS_STRIDE, wm * 32 + 16, 0, lane);
    #pragma unroll
    for (int ni = 0; ni < 4; ++ni) {
      v16bf bfv = frag_b(Bs, LDS_STRIDE, wn * 64 + ni * 16, 0, lane);
      acc[0][ni] = wmma_bf16(a0, bfv, acc[0][ni]);
      acc[1][ni] = wmma_bf16(a1, bfv, acc[1][ni]);
    }
  }
#endif

  // C/D layout: lane = N%16 + 16*(M_half); VGPR i holds M = 8*M_half + i.
  int nn = lane & 15;
  int mh = ((lane >> 4) & 1) * 8;
  #pragma unroll
  for (int mi = 0; mi < 2; ++mi) {
    #pragma unroll
    for (int ni = 0; ni < 4; ++ni) {
      v8f a = acc[mi][ni];
      int n = n0 + wn * 64 + ni * 16 + nn;
      #pragma unroll
      for (int i = 0; i < 8; ++i) {
        int m = m0 + wm * 32 + mi * 16 + mh + i;
        float v = a[i];
        if (mode == 1)      v = softplus_stable(v + bias[n]);
        else if (mode == 2) v += resid[(size_t)m * ldc + n];
        C[(size_t)m * ldc + n] = v;
      }
    }
  }
}

// ---- Small GEMM (x_proj, N=96): one wave, 32x32 tile, direct loads. ----
__global__ void __launch_bounds__(32)
gemm_bf16_kernel(const unsigned short* __restrict__ A, int lda,
                 const unsigned short* __restrict__ B, int ldb,
                 float* __restrict__ C, int ldc, int K) {
  int lane = threadIdx.x;
  int n0 = blockIdx.x * 32;
  int m0 = blockIdx.y * 32;
  v8f acc00 = {}, acc01 = {}, acc10 = {}, acc11 = {};
  for (int k0 = 0; k0 < K; k0 += 32) {
    v16bf a0 = frag_a(A, lda, m0,      k0, lane);
    v16bf a1 = frag_a(A, lda, m0 + 16, k0, lane);
    v16bf b0 = frag_b(B, ldb, n0,      k0, lane);
    v16bf b1 = frag_b(B, ldb, n0 + 16, k0, lane);
    acc00 = wmma_bf16(a0, b0, acc00);
    acc01 = wmma_bf16(a0, b1, acc01);
    acc10 = wmma_bf16(a1, b0, acc10);
    acc11 = wmma_bf16(a1, b1, acc11);
  }
  int nn = lane & 15;
  int mh = ((lane >> 4) & 1) * 8;
  v8f accs[2][2] = {acc00, acc01, acc10, acc11};
  #pragma unroll
  for (int mi = 0; mi < 2; ++mi) {
    #pragma unroll
    for (int ni = 0; ni < 2; ++ni) {
      v8f a = accs[mi][ni];
      int n = n0 + ni * 16 + nn;
      #pragma unroll
      for (int i = 0; i < 8; ++i) {
        int m = m0 + mi * 16 + mh + i;
        C[(size_t)m * ldc + n] = a[i];
      }
    }
  }
}

// Depthwise causal conv1d (kernel 4) + bias + SiLU, bf16 output.
__global__ void conv_silu_kernel(const float* __restrict__ xz,   // [BT,4096]
                                 const float* __restrict__ cw,   // [D_INNER,4]
                                 const float* __restrict__ cb,
                                 unsigned short* __restrict__ u_bf) {
  int gid = blockIdx.x * blockDim.x + threadIdx.x;
  if (gid >= BT * D_INNER) return;
  int d = gid & (D_INNER - 1);
  int r = gid >> 11;
  int t = r & (SEQ - 1);
  float acc = cb[d];
  #pragma unroll
  for (int j = 0; j < D_CONV; ++j) {
    int tt = t - (D_CONV - 1) + j;
    if (tt >= 0)
      acc += cw[d * D_CONV + j] * xz[(size_t)(r - (D_CONV - 1) + j) * (2 * D_INNER) + d];
  }
  u_bf[gid] = f32_to_bf16_bits(silu(acc));
}

// Extract dt-rank columns of x_dbl (fp32 [BT,96]) into bf16 [BT,64].
__global__ void dt_extract_kernel(const float* __restrict__ x_dbl,
                                  unsigned short* __restrict__ dtr) {
  int gid = blockIdx.x * blockDim.x + threadIdx.x;
  if (gid >= BT * DT_RANK) return;
  int r = gid >> 6, c = gid & 63;
  dtr[gid] = f32_to_bf16_bits(x_dbl[(size_t)r * 96 + c]);
}

// Selective scan + silu(z) gating. 16 lanes per (b,d) channel, one state/lane.
__global__ void scan_gate_kernel(const float* __restrict__ dt,     // [BT,D_INNER]
                                 const unsigned short* __restrict__ u_bf,
                                 const float* __restrict__ x_dbl,  // [BT,96]
                                 const float* __restrict__ xz,     // [BT,4096]
                                 const float* __restrict__ A_log,  // [D_INNER,16]
                                 const float* __restrict__ Dvec,   // [D_INNER]
                                 unsigned short* __restrict__ y_bf) {
  int gid = blockIdx.x * blockDim.x + threadIdx.x;
  int ch = gid >> 4;
  int s  = gid & 15;
  if (ch >= BATCH * D_INNER) return;
  int b = ch >> 11;
  int d = ch & (D_INNER - 1);

  float Ads = -__expf(A_log[d * D_STATE + s]);
  float Dd  = Dvec[d];
  float h   = 0.0f;
  int rbase = b * SEQ;

  for (int t = 0; t < SEQ; ++t) {
    int r = rbase + t;
    float dtv = dt[(size_t)r * D_INNER + d];
    float uv  = bf16_bits_to_f32(u_bf[(size_t)r * D_INNER + d]);
    float Bv  = x_dbl[(size_t)r * 96 + DT_RANK + s];
    float Cv  = x_dbl[(size_t)r * 96 + DT_RANK + D_STATE + s];
    float dA  = __expf(dtv * Ads);
    h = dA * h + dtv * Bv * uv;
    float y = h * Cv;
    y += __shfl_xor(y, 1, 32);
    y += __shfl_xor(y, 2, 32);
    y += __shfl_xor(y, 4, 32);
    y += __shfl_xor(y, 8, 32);
    if (s == 0) {
      float z  = xz[(size_t)r * (2 * D_INNER) + D_INNER + d];
      float yy = (y + uv * Dd) * silu(z);
      y_bf[(size_t)r * D_INNER + d] = f32_to_bf16_bits(yy);
    }
  }
}

// ------------------------------ launcher -----------------------------------

extern "C" void kernel_launch(void* const* d_in, const int* in_sizes, int n_in,
                              void* d_out, int out_size, void* d_ws, size_t ws_size,
                              hipStream_t stream) {
  const float* x        = (const float*)d_in[0];   // [2,1024,1024]
  const float* norm_w   = (const float*)d_in[1];
  const float* norm_b   = (const float*)d_in[2];
  const float* in_w     = (const float*)d_in[3];   // [4096,1024]
  const float* conv_w   = (const float*)d_in[4];   // [2048,1,4]
  const float* conv_b   = (const float*)d_in[5];
  const float* xp_w     = (const float*)d_in[6];   // [96,2048]
  const float* dt_w     = (const float*)d_in[7];   // [2048,64]
  const float* dt_b     = (const float*)d_in[8];
  const float* A_log    = (const float*)d_in[9];   // [2048,16]
  const float* Dvec     = (const float*)d_in[10];
  const float* out_w    = (const float*)d_in[11];  // [1024,2048]
  float* out            = (float*)d_out;           // [2,1024,1024]

  char* wsb = (char*)d_ws;
  size_t off = 0;
  auto wsalloc = [&](size_t bytes) -> void* {
    void* p = wsb + off;
    off = (off + bytes + 255) & ~(size_t)255;
    return p;
  };
  unsigned short* xn_bf   = (unsigned short*)wsalloc((size_t)BT * D_MODEL * 2);
  unsigned short* w_in_bf = (unsigned short*)wsalloc((size_t)2 * D_INNER * D_MODEL * 2);
  unsigned short* w_xp_bf = (unsigned short*)wsalloc((size_t)96 * D_INNER * 2);
  unsigned short* w_dt_bf = (unsigned short*)wsalloc((size_t)D_INNER * DT_RANK * 2);
  unsigned short* w_out_bf= (unsigned short*)wsalloc((size_t)D_MODEL * D_INNER * 2);
  float*          xz      = (float*)wsalloc((size_t)BT * 2 * D_INNER * 4);
  unsigned short* u_bf    = (unsigned short*)wsalloc((size_t)BT * D_INNER * 2);
  float*          x_dbl   = (float*)wsalloc((size_t)BT * 96 * 4);
  unsigned short* dtr_bf  = (unsigned short*)wsalloc((size_t)BT * DT_RANK * 2);
  float*          dt_full = (float*)wsalloc((size_t)BT * D_INNER * 4);
  unsigned short* y_bf    = (unsigned short*)wsalloc((size_t)BT * D_INNER * 2);

  // Weight fp32 -> bf16 conversions.
  {
    int n;
    n = 2 * D_INNER * D_MODEL;
    cvt_f32_bf16_kernel<<<(n + 255) / 256, 256, 0, stream>>>(in_w, w_in_bf, n);
    n = 96 * D_INNER;
    cvt_f32_bf16_kernel<<<(n + 255) / 256, 256, 0, stream>>>(xp_w, w_xp_bf, n);
    n = D_INNER * DT_RANK;
    cvt_f32_bf16_kernel<<<(n + 255) / 256, 256, 0, stream>>>(dt_w, w_dt_bf, n);
    n = D_MODEL * D_INNER;
    cvt_f32_bf16_kernel<<<(n + 255) / 256, 256, 0, stream>>>(out_w, w_out_bf, n);
  }

  // 1) LayerNorm -> bf16
  ln_kernel<<<BT, 256, 0, stream>>>(x, norm_w, norm_b, xn_bf);

  // 2) in_proj: [2048,1024] x [4096,1024]^T -> xz fp32 [2048,4096]
  gemm_bf16_lds_kernel<<<dim3((2 * D_INNER) / 128, BT / 128), 256, 0, stream>>>(
      xn_bf, D_MODEL, w_in_bf, D_MODEL, xz, 2 * D_INNER, D_MODEL, 0, nullptr, nullptr);

  // 3) depthwise causal conv + SiLU -> u bf16 [2048,2048]
  {
    int n = BT * D_INNER;
    conv_silu_kernel<<<(n + 255) / 256, 256, 0, stream>>>(xz, conv_w, conv_b, u_bf);
  }

  // 4) x_proj: [2048,2048] x [96,2048]^T -> x_dbl fp32 [2048,96]
  gemm_bf16_kernel<<<dim3(96 / 32, BT / 32), 32, 0, stream>>>(
      u_bf, D_INNER, w_xp_bf, D_INNER, x_dbl, 96, D_INNER);

  // 5) dt low-rank slice -> bf16
  {
    int n = BT * DT_RANK;
    dt_extract_kernel<<<(n + 255) / 256, 256, 0, stream>>>(x_dbl, dtr_bf);
  }

  // 6) dt_proj + bias + softplus: [2048,64] x [2048,64]^T -> dt fp32
  gemm_bf16_lds_kernel<<<dim3(D_INNER / 128, BT / 128), 256, 0, stream>>>(
      dtr_bf, DT_RANK, w_dt_bf, DT_RANK, dt_full, D_INNER, DT_RANK, 1, dt_b, nullptr);

  // 7) selective scan + gating -> y bf16 [2048,2048]
  {
    int n = BATCH * D_INNER * D_STATE;   // 65536 threads
    scan_gate_kernel<<<(n + 255) / 256, 256, 0, stream>>>(
        dt_full, u_bf, x_dbl, xz, A_log, Dvec, y_bf);
  }

  // 8) out_proj + residual: [2048,2048] x [1024,2048]^T + x -> out fp32
  gemm_bf16_lds_kernel<<<dim3(D_MODEL / 128, BT / 128), 256, 0, stream>>>(
      y_bf, D_INNER, w_out_bf, D_MODEL, out, D_MODEL, D_INNER, 2, nullptr, x);

  (void)in_sizes; (void)n_in; (void)out_size; (void)ws_size;
}
